// TopKMoE_21199958573496
// MI455X (gfx1250) — matrix operand order repaired and checked by reading, not compiled, not run on hardware
//
#include <hip/hip_runtime.h>
#include <math.h>

// Problem sizes (fixed by the reference)
#define Bb 2
#define Ll 2048
#define Hh 1024
#define Ii 4096
#define Ee 8
#define Kk 2
#define Nn (Bb * Ll)   // 4096 tokens

typedef __attribute__((ext_vector_type(16))) __bf16 v16bf;
typedef __attribute__((ext_vector_type(8)))  float  v8f;
typedef __attribute__((ext_vector_type(4)))  unsigned int u32x4;

union FragBF { v16bf v; u32x4 q[2]; };
union PackBF { u32x4 q; __bf16 b[8]; };

// ---------------------------------------------------------------------------
// CDNA5 async copy: global -> LDS, 16B per lane, tracked by ASYNCcnt.
// INST_OFFSET is added to BOTH the per-lane LDS offset and the global address.
// ---------------------------------------------------------------------------
__device__ __forceinline__ void async_copy_b128(unsigned lds_off, const void* gaddr) {
    asm volatile("global_load_async_to_lds_b128 %0, %1, off"
                 :: "v"(lds_off), "v"(gaddr) : "memory");
}
__device__ __forceinline__ void async_copy_b128_o64(unsigned lds_off, const void* gaddr) {
    asm volatile("global_load_async_to_lds_b128 %0, %1, off offset:64"
                 :: "v"(lds_off), "v"(gaddr) : "memory");
}
__device__ __forceinline__ void wait_async() {
    asm volatile("s_wait_asynccnt 0x0" ::: "memory");
}

// ---------------------------------------------------------------------------
// Kernel 0: zero per-expert counters (ws is poisoned, must init each call)
// ---------------------------------------------------------------------------
__global__ void moe_zero_kernel(int* __restrict__ cnt) {
    if (threadIdx.x < Ee) cnt[threadIdx.x] = 0;
}

// ---------------------------------------------------------------------------
// Kernel 1: gating. One block per token.
// ---------------------------------------------------------------------------
__global__ void moe_gate_kernel(const float* __restrict__ x,
                                const float* __restrict__ Wg,
                                const float* __restrict__ bg,
                                __bf16* __restrict__ xb,
                                int* __restrict__ cnt,
                                int* __restrict__ list,
                                int* __restrict__ tok_e,
                                int* __restrict__ tok_pos,
                                float* __restrict__ tok_w) {
    __shared__ float xs[Hh];
    __shared__ float logit[Ee];
    const int n = blockIdx.x;
    const int t = threadIdx.x;          // 256 threads = 8 waves
    const float* xr = x + (size_t)n * Hh;

    for (int i = t; i < Hh; i += 256) {
        float v = xr[i];
        xs[i] = v;
        xb[(size_t)n * Hh + i] = (__bf16)v;
    }
    __syncthreads();

    const int wave = t >> 5, lane = t & 31;
    {
        float acc = 0.f;
        for (int i = lane; i < Hh; i += 32)
            acc += xs[i] * Wg[i * Ee + wave];
        for (int off = 16; off > 0; off >>= 1)
            acc += __shfl_down(acc, off, 32);
        if (lane == 0) logit[wave] = acc + bg[wave];
    }
    __syncthreads();

    if (t == 0) {
        int i0 = 0; float v0 = logit[0];
        #pragma unroll
        for (int e = 1; e < Ee; ++e) if (logit[e] > v0) { v0 = logit[e]; i0 = e; }
        int i1 = (i0 == 0) ? 1 : 0; float v1 = -3.4e38f;
        #pragma unroll
        for (int e = 0; e < Ee; ++e) if (e != i0 && logit[e] > v1) { v1 = logit[e]; i1 = e; }
        float e0 = 1.0f, e1 = __expf(v1 - v0);   // softmax over two winners, v0 max
        float s  = e0 + e1;
        int p0 = atomicAdd(&cnt[i0], 1);
        int p1 = atomicAdd(&cnt[i1], 1);
        list[i0 * Nn + p0] = n;
        list[i1 * Nn + p1] = n;
        tok_e[n * 2 + 0] = i0;  tok_e[n * 2 + 1] = i1;
        tok_pos[n * 2 + 0] = p0; tok_pos[n * 2 + 1] = p1;
        tok_w[n * 2 + 0] = e0 / s; tok_w[n * 2 + 1] = e1 / s;
    }
}

// ---------------------------------------------------------------------------
// Kernel 2: tiny exclusive scan over E=8 counts -> packed row bases
// ---------------------------------------------------------------------------
__global__ void moe_scan_kernel(const int* __restrict__ cnt, int* __restrict__ base) {
    if (threadIdx.x == 0) {
        int s = 0;
        #pragma unroll
        for (int e = 0; e < Ee; ++e) { base[e] = s; s += cnt[e]; }
    }
}

// ---------------------------------------------------------------------------
// Grouped GEMM1: h = gelu(X_e * W1_e + b1_e), gathered token rows.
// 256 threads (8 waves), tile 64(M) x 128(N), K-step 64, double-buffered LDS,
// software-pipelined: next A via async-to-LDS + next B global loads issued
// before the current tile's 8 WMMAs per wave.
// ---------------------------------------------------------------------------
__global__ void moe_ffn1_kernel(const __bf16* __restrict__ xb,
                                const float* __restrict__ W1,
                                const float* __restrict__ b1,
                                const int* __restrict__ cnt,
                                const int* __restrict__ base,
                                const int* __restrict__ list,
                                __bf16* __restrict__ h) {
    const int e  = blockIdx.z;
    const int m0 = blockIdx.y * 64;
    const int n0 = blockIdx.x * 128;
    const int count = cnt[e];
    if (m0 >= count) return;

    __shared__ __bf16 As[2][64][72];    // [buf][row][k]
    __shared__ __bf16 Bs[2][128][72];   // [buf][col][k]
    __shared__ int rowTok[64];

    const int t = threadIdx.x;
    if (t < 64) {
        int r = m0 + t;
        rowTok[t] = (r < count) ? list[e * Nn + r] : -1;
        if (r >= count) {               // zero-fill invalid rows in both buffers
            #pragma unroll
            for (int kc = 0; kc < 64; kc += 8) {
                u32x4 z = {};
                *(u32x4*)(&As[0][t][kc]) = z;
                *(u32x4*)(&As[1][t][kc]) = z;
            }
        }
    }
    __syncthreads();

    const int wave = t >> 5, lane = t & 31;
    const int wr = (wave & 3) * 16;     // wave row offset (4 x 16 = 64)
    const int wc = (wave >> 2) * 64;    // wave col offset (2 x 64 = 128)

    v8f acc[4] = {{}, {}, {}, {}};
    const float* W1e = W1 + (size_t)e * Hh * Ii;

    // loop-invariant staging coordinates
    const int ar = t >> 2, akc = (t & 3) * 8;          // A: 2x16B per thread
    const unsigned aoff0 = (unsigned)(size_t)&As[0][ar][akc];
    const unsigned aoff1 = (unsigned)(size_t)&As[1][ar][akc];
    const int tok = rowTok[ar];
    const __bf16* asrc = (tok >= 0) ? (xb + (size_t)tok * Hh + akc) : nullptr;
    const int bc2 = (t & 63) * 2, bkh = (t >> 6) * 16; // B: 2 cols x 16 k
    const float* bsrc = W1e + (size_t)bkh * Ii + n0 + bc2;

    // --- prologue: stage tile 0 into buffer 0
    if (tok >= 0) {
        async_copy_b128(aoff0, asrc);
        async_copy_b128_o64(aoff0, asrc);
    }
    {
        PackBF pa0, pa1, pb0, pb1;
        #pragma unroll
        for (int j = 0; j < 8; ++j) {
            float2 f0 = *(const float2*)(bsrc + (size_t)j * Ii);
            float2 f1 = *(const float2*)(bsrc + (size_t)(j + 8) * Ii);
            pa0.b[j] = (__bf16)f0.x; pb0.b[j] = (__bf16)f0.y;
            pa1.b[j] = (__bf16)f1.x; pb1.b[j] = (__bf16)f1.y;
        }
        *(u32x4*)(&Bs[0][bc2 + 0][bkh])     = pa0.q;
        *(u32x4*)(&Bs[0][bc2 + 0][bkh + 8]) = pa1.q;
        *(u32x4*)(&Bs[0][bc2 + 1][bkh])     = pb0.q;
        *(u32x4*)(&Bs[0][bc2 + 1][bkh + 8]) = pb1.q;
    }
    wait_async();
    __syncthreads();

    const int arow  = wr + (lane & 15);
    const int ahalf = (lane >> 4) * 16;
    const int bcol  = lane & 15;
    const int fkh   = (lane >> 4) * 16;

    const int NIT = Hh / 64;
    for (int it = 0; it < NIT; ++it) {
        const int cur = it & 1, nxt = cur ^ 1;
        const bool hasNext = (it + 1) < NIT;
        const int k0n = (it + 1) * 64;

        // --- issue next tile's global B loads (held in registers over compute)
        float2 f0[8], f1[8];
        if (hasNext) {
            const float* bp = bsrc + (size_t)k0n * Ii;
            #pragma unroll
            for (int j = 0; j < 8; ++j) {
                f0[j] = *(const float2*)(bp + (size_t)j * Ii);
                f1[j] = *(const float2*)(bp + (size_t)(j + 8) * Ii);
            }
            // --- issue next tile's async A copies
            if (tok >= 0) {
                unsigned ao = nxt ? aoff1 : aoff0;
                async_copy_b128(ao, asrc + k0n);
                async_copy_b128_o64(ao, asrc + k0n);
            }
        }

        // --- compute current tile: 2 K-substeps x 4 col subtiles = 8 WMMAs
        #pragma unroll
        for (int s = 0; s < 2; ++s) {
            FragBF a;
            a.q[0] = *(const u32x4*)(&As[cur][arow][s * 32 + ahalf]);
            a.q[1] = *(const u32x4*)(&As[cur][arow][s * 32 + ahalf + 8]);
            #pragma unroll
            for (int s2 = 0; s2 < 4; ++s2) {
                FragBF b;
                b.q[0] = *(const u32x4*)(&Bs[cur][wc + s2 * 16 + bcol][s * 32 + fkh]);
                b.q[1] = *(const u32x4*)(&Bs[cur][wc + s2 * 16 + bcol][s * 32 + fkh + 8]);
                acc[s2] = __builtin_amdgcn_wmma_f32_16x16x32_bf16(
                    false, a.v, false, b.v, (short)0, acc[s2], false, false);
            }
        }

        // --- convert + store next B tile, fence, flip
        if (hasNext) {
            PackBF pa0, pa1, pb0, pb1;
            #pragma unroll
            for (int j = 0; j < 8; ++j) {
                pa0.b[j] = (__bf16)f0[j].x; pb0.b[j] = (__bf16)f0[j].y;
                pa1.b[j] = (__bf16)f1[j].x; pb1.b[j] = (__bf16)f1[j].y;
            }
            *(u32x4*)(&Bs[nxt][bc2 + 0][bkh])     = pa0.q;
            *(u32x4*)(&Bs[nxt][bc2 + 0][bkh + 8]) = pa1.q;
            *(u32x4*)(&Bs[nxt][bc2 + 1][bkh])     = pb0.q;
            *(u32x4*)(&Bs[nxt][bc2 + 1][bkh + 8]) = pb1.q;
        }
        wait_async();
        __syncthreads();
    }

    // --- epilogue: bias + exact GELU, packed bf16 h rows
    const int baseE = base[e];
    const int Nl = lane & 15, Mb = (lane >> 4) * 8;
    #pragma unroll
    for (int s2 = 0; s2 < 4; ++s2) {
        #pragma unroll
        for (int j = 0; j < 8; ++j) {
            int r = m0 + wr + Mb + j;
            if (r < count) {
                int col = n0 + wc + s2 * 16 + Nl;
                float v = acc[s2][j] + b1[e * Ii + col];
                float g = 0.5f * v * (1.0f + erff(v * 0.70710678f));
                h[(size_t)(baseE + r) * Ii + col] = (__bf16)g;
            }
        }
    }
}

// ---------------------------------------------------------------------------
// Grouped GEMM2: contrib = h_e * W2_e + b2_e (rows already packed per expert)
// ---------------------------------------------------------------------------
__global__ void moe_ffn2_kernel(const __bf16* __restrict__ h,
                                const float* __restrict__ W2,
                                const float* __restrict__ b2,
                                const int* __restrict__ cnt,
                                const int* __restrict__ base,
                                float* __restrict__ contrib) {
    const int e  = blockIdx.z;
    const int m0 = blockIdx.y * 64;
    const int n0 = blockIdx.x * 128;
    const int count = cnt[e];
    if (m0 >= count) return;
    const int baseE = base[e];

    __shared__ __bf16 As[2][64][72];
    __shared__ __bf16 Bs[2][128][72];

    const int t = threadIdx.x;
    if (t < 64 && (m0 + t) >= count) {
        #pragma unroll
        for (int kc = 0; kc < 64; kc += 8) {
            u32x4 z = {};
            *(u32x4*)(&As[0][t][kc]) = z;
            *(u32x4*)(&As[1][t][kc]) = z;
        }
    }
    __syncthreads();

    const int wave = t >> 5, lane = t & 31;
    const int wr = (wave & 3) * 16;
    const int wc = (wave >> 2) * 64;

    v8f acc[4] = {{}, {}, {}, {}};
    const float* W2e = W2 + (size_t)e * Ii * Hh;

    const int ar = t >> 2, akc = (t & 3) * 8;
    const unsigned aoff0 = (unsigned)(size_t)&As[0][ar][akc];
    const unsigned aoff1 = (unsigned)(size_t)&As[1][ar][akc];
    const bool avalid = (m0 + ar) < count;
    const __bf16* asrc = h + (size_t)(baseE + m0 + ar) * Ii + akc;
    const int bc2 = (t & 63) * 2, bkh = (t >> 6) * 16;
    const float* bsrc = W2e + (size_t)bkh * Hh + n0 + bc2;

    if (avalid) {
        async_copy_b128(aoff0, asrc);
        async_copy_b128_o64(aoff0, asrc);
    }
    {
        PackBF pa0, pa1, pb0, pb1;
        #pragma unroll
        for (int j = 0; j < 8; ++j) {
            float2 g0 = *(const float2*)(bsrc + (size_t)j * Hh);
            float2 g1 = *(const float2*)(bsrc + (size_t)(j + 8) * Hh);
            pa0.b[j] = (__bf16)g0.x; pb0.b[j] = (__bf16)g0.y;
            pa1.b[j] = (__bf16)g1.x; pb1.b[j] = (__bf16)g1.y;
        }
        *(u32x4*)(&Bs[0][bc2 + 0][bkh])     = pa0.q;
        *(u32x4*)(&Bs[0][bc2 + 0][bkh + 8]) = pa1.q;
        *(u32x4*)(&Bs[0][bc2 + 1][bkh])     = pb0.q;
        *(u32x4*)(&Bs[0][bc2 + 1][bkh + 8]) = pb1.q;
    }
    wait_async();
    __syncthreads();

    const int arow  = wr + (lane & 15);
    const int ahalf = (lane >> 4) * 16;
    const int bcol  = lane & 15;
    const int fkh   = (lane >> 4) * 16;

    const int NIT = Ii / 64;
    for (int it = 0; it < NIT; ++it) {
        const int cur = it & 1, nxt = cur ^ 1;
        const bool hasNext = (it + 1) < NIT;
        const int k0n = (it + 1) * 64;

        float2 f0[8], f1[8];
        if (hasNext) {
            const float* bp = bsrc + (size_t)k0n * Hh;
            #pragma unroll
            for (int j = 0; j < 8; ++j) {
                f0[j] = *(const float2*)(bp + (size_t)j * Hh);
                f1[j] = *(const float2*)(bp + (size_t)(j + 8) * Hh);
            }
            if (avalid) {
                unsigned ao = nxt ? aoff1 : aoff0;
                async_copy_b128(ao, asrc + k0n);
                async_copy_b128_o64(ao, asrc + k0n);
            }
        }

        #pragma unroll
        for (int s = 0; s < 2; ++s) {
            FragBF a;
            a.q[0] = *(const u32x4*)(&As[cur][arow][s * 32 + ahalf]);
            a.q[1] = *(const u32x4*)(&As[cur][arow][s * 32 + ahalf + 8]);
            #pragma unroll
            for (int s2 = 0; s2 < 4; ++s2) {
                FragBF b;
                b.q[0] = *(const u32x4*)(&Bs[cur][wc + s2 * 16 + bcol][s * 32 + fkh]);
                b.q[1] = *(const u32x4*)(&Bs[cur][wc + s2 * 16 + bcol][s * 32 + fkh + 8]);
                acc[s2] = __builtin_amdgcn_wmma_f32_16x16x32_bf16(
                    false, a.v, false, b.v, (short)0, acc[s2], false, false);
            }
        }

        if (hasNext) {
            PackBF pa0, pa1, pb0, pb1;
            #pragma unroll
            for (int j = 0; j < 8; ++j) {
                pa0.b[j] = (__bf16)f0[j].x; pb0.b[j] = (__bf16)f0[j].y;
                pa1.b[j] = (__bf16)f1[j].x; pb1.b[j] = (__bf16)f1[j].y;
            }
            *(u32x4*)(&Bs[nxt][bc2 + 0][bkh])     = pa0.q;
            *(u32x4*)(&Bs[nxt][bc2 + 0][bkh + 8]) = pa1.q;
            *(u32x4*)(&Bs[nxt][bc2 + 1][bkh])     = pb0.q;
            *(u32x4*)(&Bs[nxt][bc2 + 1][bkh + 8]) = pb1.q;
        }
        wait_async();
        __syncthreads();
    }

    const int Nl = lane & 15, Mb = (lane >> 4) * 8;
    #pragma unroll
    for (int s2 = 0; s2 < 4; ++s2) {
        #pragma unroll
        for (int j = 0; j < 8; ++j) {
            int r = m0 + wr + Mb + j;
            if (r < count) {
                int col = n0 + wc + s2 * 16 + Nl;
                contrib[(size_t)(baseE + r) * Hh + col] = acc[s2][j] + b2[e * Hh + col];
            }
        }
    }
}

// ---------------------------------------------------------------------------
// Kernel 5: weighted combine of the token's two expert contributions
// ---------------------------------------------------------------------------
__global__ void moe_combine_kernel(const float* __restrict__ contrib,
                                   const int* __restrict__ base,
                                   const int* __restrict__ tok_e,
                                   const int* __restrict__ tok_pos,
                                   const float* __restrict__ tok_w,
                                   float* __restrict__ out) {
    const int n = blockIdx.x;
    const int r0 = base[tok_e[n * 2 + 0]] + tok_pos[n * 2 + 0];
    const int r1 = base[tok_e[n * 2 + 1]] + tok_pos[n * 2 + 1];
    const float w0 = tok_w[n * 2 + 0], w1 = tok_w[n * 2 + 1];
    const float* c0 = contrib + (size_t)r0 * Hh;
    const float* c1 = contrib + (size_t)r1 * Hh;
    float* o = out + (size_t)n * Hh;
    for (int i = threadIdx.x; i < Hh; i += 256)
        o[i] = w0 * c0[i] + w1 * c1[i];
}

// ---------------------------------------------------------------------------
extern "C" void kernel_launch(void* const* d_in, const int* in_sizes, int n_in,
                              void* d_out, int out_size, void* d_ws, size_t ws_size,
                              hipStream_t stream) {
    const float* x  = (const float*)d_in[0];
    const float* Wg = (const float*)d_in[1];
    const float* bg = (const float*)d_in[2];
    const float* W1 = (const float*)d_in[3];
    const float* b1 = (const float*)d_in[4];
    const float* W2 = (const float*)d_in[5];
    const float* b2 = (const float*)d_in[6];
    float* out = (float*)d_out;

    // Workspace carve-up (256B aligned slices)
    char* ws = (char*)d_ws;
    size_t off = 0;
    auto take = [&](size_t bytes) {
        char* p = ws + off;
        off += (bytes + 255) & ~(size_t)255;
        return p;
    };
    __bf16* xb      = (__bf16*)take((size_t)Nn * Hh * sizeof(__bf16));        // 8 MB
    int*    cnt     = (int*)   take(Ee * sizeof(int));
    int*    basep   = (int*)   take(Ee * sizeof(int));
    int*    list    = (int*)   take((size_t)Ee * Nn * sizeof(int));           // 128 KB
    int*    tok_e   = (int*)   take((size_t)Nn * Kk * sizeof(int));
    int*    tok_pos = (int*)   take((size_t)Nn * Kk * sizeof(int));
    float*  tok_w   = (float*) take((size_t)Nn * Kk * sizeof(float));
    __bf16* hbuf    = (__bf16*)take((size_t)Nn * Kk * Ii * sizeof(__bf16));   // 64 MB
    float*  contrib = (float*) take((size_t)Nn * Kk * Hh * sizeof(float));    // 32 MB

    moe_zero_kernel<<<1, 32, 0, stream>>>(cnt);
    moe_gate_kernel<<<Nn, 256, 0, stream>>>(x, Wg, bg, xb, cnt, list, tok_e, tok_pos, tok_w);
    moe_scan_kernel<<<1, 32, 0, stream>>>(cnt, basep);
    moe_ffn1_kernel<<<dim3(Ii / 128, Nn / 64, Ee), 256, 0, stream>>>(xb, W1, b1, cnt, basep, list, hbuf);
    moe_ffn2_kernel<<<dim3(Hh / 128, Nn / 64, Ee), 256, 0, stream>>>(hbuf, W2, b2, cnt, basep, contrib);
    moe_combine_kernel<<<Nn, 256, 0, stream>>>(contrib, basep, tok_e, tok_pos, tok_w, out);
}
